// DecoderLSTM_4922032521775
// MI455X (gfx1250) — matrix operand order repaired and checked by reading, not compiled, not run on hardware
//
#include <hip/hip_runtime.h>
#include <hip/hip_bf16.h>

// Problem constants (match reference).
#define B_  16
#define F_  512
#define V_  8192
#define G4_ 32768   // 4*V
#define ML_ 20

// GEMM tiling.
#define KC_    256                   // K elements per TDM chunk (bf16)
#define ROWS_  64                    // W rows per block (4 waves x 16)
#define PADB_  16                    // LDS pad bytes per row (via TDM pad)
#define LROW_  (KC_ * 2 + PADB_)     // 528 B LDS row stride
#define TILEB_ (ROWS_ * LROW_)       // 33792 B per buffer

typedef __attribute__((ext_vector_type(16))) __bf16 v16bf;
typedef __attribute__((ext_vector_type(8)))  __bf16 v8bf;
typedef __attribute__((ext_vector_type(4)))  __bf16 v4bf;
typedef __attribute__((ext_vector_type(8)))  float  v8f;
typedef __attribute__((ext_vector_type(4)))  float  v4f;
typedef __attribute__((ext_vector_type(4)))  unsigned int u32x4;
typedef __attribute__((ext_vector_type(8)))  unsigned int u32x8;

union AFrag { v16bf v; v8bf h[2]; };

__device__ __forceinline__ float sigm_(float x) { return 1.0f / (1.0f + __expf(-x)); }

// ---------------------------------------------------------------------------
// One-time setup kernels
// ---------------------------------------------------------------------------
__global__ void cvt_f32_to_bf16_x4(const v4f* __restrict__ src,
                                   v4bf* __restrict__ dst, long n4) {
    long i = (long)blockIdx.x * blockDim.x + threadIdx.x;
    if (i < n4) {
        v4f x = src[i];
        dst[i] = __builtin_convertvector(x, v4bf);
    }
}

__global__ void bias_add(const float* __restrict__ b_ih,
                         const float* __restrict__ b_hh,
                         float* __restrict__ bias, int n) {
    int i = blockIdx.x * blockDim.x + threadIdx.x;
    if (i < n) bias[i] = b_ih[i] + b_hh[i];
}

__global__ void zero_f32(float* __restrict__ p, int n) {
    int i = blockIdx.x * blockDim.x + threadIdx.x;
    if (i < n) p[i] = 0.0f;
}

// ---------------------------------------------------------------------------
// TDM issue: 2-D tile (ROWS_ x KC_ bf16) from row-major W into LDS,
// with per-row padding so LDS row stride = LROW_ bytes.
// D# group 0/1 per CDNA5 ISA 8.3/8.4.
// ---------------------------------------------------------------------------
__device__ __forceinline__ void tdm_issue_tile(const __bf16* gsrc, unsigned ldsOff,
                                               unsigned tensorD0, unsigned tensorD1,
                                               unsigned stride) {
    unsigned long long ga = (unsigned long long)(uintptr_t)gsrc;
    u32x4 g0;
    g0[0] = 1u;                                   // count=1, user descriptor
    g0[1] = ldsOff;                               // lds_addr (bytes)
    g0[2] = (unsigned)ga;                         // global_addr[31:0]
    g0[3] = (unsigned)((ga >> 32) & 0x01FFFFFFu)  // global_addr[56:32]
            | (2u << 30);                         // type = 2 ("image")
    u32x8 g1;
    // data_size=1 (2B) | pad_enable | pad_interval=6 (128 DW = 512 B)
    // | pad_amount=3 (4 DW = 16 B)
    g1[0] = (1u << 16) | (1u << 20) | (6u << 22) | (3u << 25);
    g1[1] = (tensorD0 & 0xFFFFu) << 16;                        // tensor_dim0[15:0]
    g1[2] = (tensorD0 >> 16) | ((tensorD1 & 0xFFFFu) << 16);   // dim0[31:16], dim1[15:0]
    g1[3] = (tensorD1 >> 16) | ((unsigned)KC_ << 16);          // dim1[31:16], tile_dim0
    g1[4] = (unsigned)ROWS_;                                   // tile_dim1 (tile_dim2=0)
    g1[5] = stride;                                            // tensor_dim0_stride[31:0]
    g1[6] = 0u;                                                // stride hi, dim1_stride lo
    g1[7] = 0u;
    asm volatile("tensor_load_to_lds %0, %1" :: "s"(g0), "s"(g1) : "memory");
}

// ---------------------------------------------------------------------------
// gates[16, 32768] = x[16,512] @ Wih^T + h[16,8192] @ Whh^T + bias
// Block = 128 threads = 4 waves; each wave owns one 16-column tile.
// Grid = 32768 / 64 = 512 blocks.
// W streamed by TDM into double-buffered LDS; A kept in registers per chunk.
// Chunks: c in [0, 2)            -> ih part (K=512)
//         c in [2, 2+32) if useH -> hh part (K=8192)
// ---------------------------------------------------------------------------
__global__ __launch_bounds__(128) void lstm_gates_wmma(
    const __bf16* __restrict__ xbf,   // [16, 512]
    const __bf16* __restrict__ hbf,   // [16, 8192]
    const __bf16* __restrict__ Wih,   // [32768, 512] row-major
    const __bf16* __restrict__ Whh,   // [32768, 8192] row-major
    const float*  __restrict__ bias,  // [32768]
    float*        __restrict__ gates, // [16, 32768]
    int useH)
{
    __shared__ __align__(128) unsigned char smem[2][TILEB_];

    const int wave   = threadIdx.x >> 5;
    const int lane   = threadIdx.x & 31;
    const int lane16 = lane & 15;
    const int laneHi = lane >> 4;
    const int nBase  = blockIdx.x * ROWS_;

    const unsigned ldsBase = (unsigned)(uintptr_t)&smem[0][0];
    // B-fragment base inside a buffer: row = wave*16 + lane16, +32B for laneHi.
    const unsigned bOff = (unsigned)(wave * 16 + lane16) * LROW_ + (unsigned)laneHi * 32u;

    const int total = 2 + (useH ? (V_ / KC_) : 0);

    // Prologue: issue chunk 0 (ih, k=0) into buffer 0.
    if (threadIdx.x < 32) {
        tdm_issue_tile(Wih + (size_t)nBase * F_, ldsBase, F_, G4_, F_);
    }

    v8f acc = {};

    for (int c = 0; c < total; ++c) {
        const int par = c & 1;

        // Per-chunk A/W parameters.
        const __bf16* Aptr;  const __bf16* Wptr;  unsigned Kdim;  unsigned kBase;
        if (c < 2) { Aptr = xbf; Wptr = Wih; Kdim = F_; kBase = (unsigned)c * KC_; }
        else       { Aptr = hbf; Wptr = Whh; Kdim = V_; kBase = (unsigned)(c - 2) * KC_; }

        // Preload A fragments for this chunk (L2-resident) before the tensor wait.
        AFrag a[KC_ / 32];
        {
            const __bf16* arow = Aptr + (size_t)lane16 * Kdim + kBase;
#pragma unroll
            for (int j = 0; j < KC_ / 32; ++j) {
                a[j].h[0] = *(const v8bf*)(arow + j * 32 + laneHi * 16);
                a[j].h[1] = *(const v8bf*)(arow + j * 32 + laneHi * 16 + 8);
            }
        }

        // Wave 0: prefetch chunk c+1, then wait for chunk c (TDM is in-order).
        if (threadIdx.x < 32) {
            if (c + 1 < total) {
                const int cn = c + 1;
                const __bf16* Wn;  unsigned Kn, kn;
                if (cn < 2) { Wn = Wih; Kn = F_; kn = (unsigned)cn * KC_; }
                else        { Wn = Whh; Kn = V_; kn = (unsigned)(cn - 2) * KC_; }
                tdm_issue_tile(Wn + (size_t)nBase * Kn + kn,
                               ldsBase + (unsigned)((cn & 1) * TILEB_), Kn, G4_, Kn);
                __builtin_amdgcn_s_wait_tensorcnt(1);
            } else {
                __builtin_amdgcn_s_wait_tensorcnt(0);
            }
        }
        __syncthreads();   // chunk c visible in LDS to all waves

        // Compute: 8 WMMAs on this chunk, B from LDS (ds_load_b128 pairs).
        const unsigned char* tb = &smem[par][0] + bOff;
#pragma unroll
        for (int j = 0; j < KC_ / 32; ++j) {
            AFrag b;
            b.h[0] = *(const v8bf*)(tb + j * 64);
            b.h[1] = *(const v8bf*)(tb + j * 64 + 16);
            acc = __builtin_amdgcn_wmma_f32_16x16x32_bf16(
                false, a[j].v, false, b.v, (short)0, acc, false, false);
        }
        __syncthreads();   // buffer 'par' free for reuse at chunk c+2
    }

    // Epilogue: add bias, scatter D layout (VGPR j -> M=j+8*laneHi, N=lane16).
    {
        const int n  = nBase + wave * 16 + lane16;
        const int m0 = laneHi * 8;
        const float bv = bias[n];
        float* gcol = gates + n;
#pragma unroll
        for (int j = 0; j < 8; ++j) {
            gcol[(size_t)(m0 + j) * G4_] = acc[j] + bv;
        }
    }
}

// ---------------------------------------------------------------------------
// LSTM cell pointwise: c,h update. idx over B*V. torch gate order: i, f, g, o.
// ---------------------------------------------------------------------------
__global__ void lstm_cell(const float* __restrict__ gates,
                          float* __restrict__ c,
                          float* __restrict__ h,
                          __bf16* __restrict__ hbf) {
    int idx = blockIdx.x * blockDim.x + threadIdx.x;   // 131072
    int b = idx >> 13;
    int v = idx & (V_ - 1);
    const float* g = gates + (size_t)b * G4_;
    float gi = sigm_(g[v]);
    float gf = sigm_(g[V_ + v]);
    float gg = tanhf(g[2 * V_ + v]);
    float go = sigm_(g[3 * V_ + v]);
    float cn = gf * c[idx] + gi * gg;
    float hn = go * tanhf(cn);
    c[idx] = cn;
    h[idx] = hn;
    hbf[idx] = (__bf16)hn;
}

// ---------------------------------------------------------------------------
// choose(): softmax over batch (axis 0) then argmax over vocab (axis 1).
// argmax_v exp(h[b,v])/D[v] == argmax_v (h[b,v] - log D[v]).
// ---------------------------------------------------------------------------
__global__ void col_logsumexp(const float* __restrict__ h,
                              float* __restrict__ logD) {
    int v = blockIdx.x * blockDim.x + threadIdx.x;   // 8192
    float s = 0.0f;
#pragma unroll
    for (int b = 0; b < B_; ++b) s += __expf(h[(size_t)b * V_ + v]);
    logD[v] = __logf(s);
}

__global__ __launch_bounds__(256) void row_argmax(const float* __restrict__ h,
                                                  const float* __restrict__ logD,
                                                  int* __restrict__ w,
                                                  int* __restrict__ outp) {
    __shared__ float sv[256];
    __shared__ int   si[256];
    const int b = blockIdx.x;
    const int t = threadIdx.x;
    float best = -3.402823466e+38f;
    int   bi   = V_;
    for (int v = t; v < V_; v += 256) {
        float s = h[(size_t)b * V_ + v] - logD[v];
        if (s > best) { best = s; bi = v; }   // v increasing -> first max kept
    }
    sv[t] = best; si[t] = bi;
    __syncthreads();
    for (int off = 128; off > 0; off >>= 1) {
        if (t < off) {
            if (sv[t + off] > sv[t] ||
                (sv[t + off] == sv[t] && si[t + off] < si[t])) {
                sv[t] = sv[t + off];
                si[t] = si[t + off];
            }
        }
        __syncthreads();
    }
    if (t == 0) { w[b] = si[0]; outp[b] = si[0]; }
}

__global__ void embed_to_x(const float* __restrict__ emb,
                           const int* __restrict__ w,
                           __bf16* __restrict__ xbf) {
    int idx = blockIdx.x * blockDim.x + threadIdx.x;   // 16*512
    int b = idx >> 9;
    int f = idx & (F_ - 1);
    xbf[idx] = (__bf16)emb[(size_t)w[b] * F_ + f];
}

// ---------------------------------------------------------------------------
// Host-side launcher
// ---------------------------------------------------------------------------
extern "C" void kernel_launch(void* const* d_in, const int* in_sizes, int n_in,
                              void* d_out, int out_size, void* d_ws, size_t ws_size,
                              hipStream_t stream) {
    const float* feature = (const float*)d_in[0];   // [16, 512]
    const float* W_ih    = (const float*)d_in[1];   // [32768, 512]
    const float* W_hh    = (const float*)d_in[2];   // [32768, 8192]
    const float* b_ih    = (const float*)d_in[3];   // [32768]
    const float* b_hh    = (const float*)d_in[4];   // [32768]
    const float* emb     = (const float*)d_in[5];   // [8192, 512]
    // d_in[6] = maxLength (device scalar); fixed to ML_ = 20 per reference.

    // Workspace carve (offsets keep >=256B alignment).
    char* ws = (char*)d_ws;
    __bf16* Whh_bf = (__bf16*)ws;  ws += (size_t)G4_ * V_ * sizeof(__bf16);  // 512 MiB
    __bf16* Wih_bf = (__bf16*)ws;  ws += (size_t)G4_ * F_ * sizeof(__bf16);  //  32 MiB
    float*  bias   = (float*)ws;   ws += (size_t)G4_ * sizeof(float);
    float*  h      = (float*)ws;   ws += (size_t)B_ * V_ * sizeof(float);
    float*  c      = (float*)ws;   ws += (size_t)B_ * V_ * sizeof(float);
    __bf16* hbf    = (__bf16*)ws;  ws += (size_t)B_ * V_ * sizeof(__bf16);
    __bf16* xbf    = (__bf16*)ws;  ws += (size_t)B_ * F_ * sizeof(__bf16);
    float*  gates  = (float*)ws;   ws += (size_t)B_ * G4_ * sizeof(float);
    float*  logD   = (float*)ws;   ws += (size_t)V_ * sizeof(float);
    int*    widx   = (int*)ws;     ws += 256;

    int* out = (int*)d_out;   // [20, 16] int tokens

    // ---- one-time setup ----
    {
        long n4 = (long)G4_ * V_ / 4;   // W_hh
        cvt_f32_to_bf16_x4<<<(int)((n4 + 255) / 256), 256, 0, stream>>>(
            (const v4f*)W_hh, (v4bf*)Whh_bf, n4);
        n4 = (long)G4_ * F_ / 4;        // W_ih
        cvt_f32_to_bf16_x4<<<(int)((n4 + 255) / 256), 256, 0, stream>>>(
            (const v4f*)W_ih, (v4bf*)Wih_bf, n4);
        n4 = (long)B_ * F_ / 4;         // feature -> x
        cvt_f32_to_bf16_x4<<<(int)((n4 + 255) / 256), 256, 0, stream>>>(
            (const v4f*)feature, (v4bf*)xbf, n4);
        bias_add<<<G4_ / 256, 256, 0, stream>>>(b_ih, b_hh, bias, G4_);
        zero_f32<<<(B_ * V_) / 256, 256, 0, stream>>>(c, B_ * V_);
    }

    // ---- 20 sequential decode steps ----
    for (int step = 0; step < ML_; ++step) {
        lstm_gates_wmma<<<G4_ / ROWS_, 128, 0, stream>>>(
            xbf, hbf, Wih_bf, Whh_bf, bias, gates, step > 0 ? 1 : 0);
        lstm_cell<<<(B_ * V_) / 256, 256, 0, stream>>>(gates, c, h, hbf);
        col_logsumexp<<<V_ / 256, 256, 0, stream>>>(h, logD);
        row_argmax<<<B_, 256, 0, stream>>>(h, logD, widx, out + step * B_);
        if (step + 1 < ML_) {
            embed_to_x<<<(B_ * F_) / 256, 256, 0, stream>>>(emb, widx, xbf);
        }
    }
}
